// AbstractTransformerBlock_80942953660908
// MI455X (gfx1250) — compile-verified, matrix operand order checked by reading, not run on hardware
//
#include <hip/hip_runtime.h>
#include <stdint.h>

typedef unsigned short u16;
typedef __attribute__((ext_vector_type(16))) __bf16 v16bf;
typedef __attribute__((ext_vector_type(8)))  float  v8f;

#define DIM    4096
#define NH     32
#define HD     128
#define HIDDEN 11008
#define SEQ    2048

union FragU { v16bf bf; uint4 u[2]; };

static __device__ inline u16 f32_to_bf16(float f) {
  unsigned u = __float_as_uint(f);
  unsigned r = u + 0x7FFFu + ((u >> 16) & 1u);   // round-to-nearest-even
  return (u16)(r >> 16);
}

// Issue one 16B global -> LDS async copy (per lane). Tracked by ASYNCcnt.
static __device__ inline void async_ld16(unsigned ldsOff, unsigned long long gaddr) {
  asm volatile("global_load_async_to_lds_b128 %0, %1, off"
               :: "v"(ldsOff), "v"(gaddr) : "memory");
}

// ---------------------------------------------------------------------------
// fp32 -> bf16 weight conversion
// ---------------------------------------------------------------------------
__global__ void cvt_bf16(const float* __restrict__ in, u16* __restrict__ out, int n) {
  int i = (blockIdx.x * blockDim.x + threadIdx.x) * 4;
  if (i + 3 < n) {
    float4 f = *(const float4*)(in + i);
    out[i]   = f32_to_bf16(f.x);
    out[i+1] = f32_to_bf16(f.y);
    out[i+2] = f32_to_bf16(f.z);
    out[i+3] = f32_to_bf16(f.w);
  }
}

// ---------------------------------------------------------------------------
// RMSNorm: f32 row -> bf16 row (scaled by w)
// ---------------------------------------------------------------------------
__global__ __launch_bounds__(256)
void rmsnorm_bf16(const float* __restrict__ x, const float* __restrict__ w,
                  u16* __restrict__ out) {
  __shared__ float red[8];
  const int row = blockIdx.x;
  const float* xr = x + (size_t)row * DIM;
  float s = 0.f;
  for (int i = threadIdx.x; i < DIM; i += 256) { float v = xr[i]; s += v * v; }
  #pragma unroll
  for (int m = 16; m >= 1; m >>= 1) s += __shfl_xor(s, m, 32);
  if ((threadIdx.x & 31) == 0) red[threadIdx.x >> 5] = s;
  __syncthreads();
  if (threadIdx.x < 8) {
    float t = red[threadIdx.x];
    #pragma unroll
    for (int m = 4; m >= 1; m >>= 1) t += __shfl_xor(t, m, 32);
    if (threadIdx.x == 0) red[0] = t;
  }
  __syncthreads();
  const float rs = rsqrtf(red[0] / (float)DIM + 1e-5f);
  u16* orow = out + (size_t)row * DIM;
  for (int i = threadIdx.x; i < DIM; i += 256)
    orow[i] = f32_to_bf16(xr[i] * rs * w[i]);
}

// ---------------------------------------------------------------------------
// NT GEMM: C[M,N] = A[M,K](bf16) * B[N,K](bf16)^T (+ optional f32 residual R)
// Block tile 128x64, K-tile 64, double-buffered LDS filled by
// GLOBAL_LOAD_ASYNC_TO_LDS_B128 (ASYNCcnt), 8 waves, 32x32 per wave,
// WMMA f32_16x16x32_bf16.
// ---------------------------------------------------------------------------
#define AT_ELEM (128 * 72)           // A tile elements (row stride 72 = 64+8 pad)
#define BT_ELEM (64 * 72)

__global__ __launch_bounds__(256)
void gemm_bf16_nt(const u16* __restrict__ A, const u16* __restrict__ B,
                  float* __restrict__ C, const float* __restrict__ R,
                  int M, int N, int K) {
  __shared__ alignas(16) u16 As[2 * AT_ELEM];
  __shared__ alignas(16) u16 Bs[2 * BT_ELEM];
  const int tid   = threadIdx.x;
  const int lane  = tid & 31;
  const int wave  = tid >> 5;
  const int wm    = wave & 3;        // 4 waves in M
  const int wn    = wave >> 2;       // 2 waves in N
  const int bm    = blockIdx.x * 128;
  const int bn    = blockIdx.y * 64;
  const int row16 = lane & 15;
  const int khalf = (lane >> 4) * 8;

  const unsigned asA = (unsigned)(uintptr_t)As;   // LDS byte offset (addr[31:0])
  const unsigned asB = (unsigned)(uintptr_t)Bs;

  v8f acc[2][2] = {};
  const int nk = K >> 6;             // K / 64

  // async prefetch of one k-tile into LDS buffer `buf`
  auto issue_tile = [&](int kt, int buf) {
    const unsigned aBase = asA + (unsigned)buf * AT_ELEM * 2;
    const unsigned bBase = asB + (unsigned)buf * BT_ELEM * 2;
    #pragma unroll
    for (int t = 0; t < 4; ++t) {                 // A: 128x64 = 1024 chunks
      int chunk = tid + t * 256;
      int r = chunk >> 3;
      int c = (chunk & 7) * 8;
      async_ld16(aBase + (unsigned)(r * 72 + c) * 2,
                 (unsigned long long)(uintptr_t)&A[(size_t)(bm + r) * K + kt * 64 + c]);
    }
    #pragma unroll
    for (int t = 0; t < 2; ++t) {                 // B: 64x64 = 512 chunks
      int chunk = tid + t * 256;
      int r = chunk >> 3;
      int c = (chunk & 7) * 8;
      async_ld16(bBase + (unsigned)(r * 72 + c) * 2,
                 (unsigned long long)(uintptr_t)&B[(size_t)(bn + r) * K + kt * 64 + c]);
    }
  };

  issue_tile(0, 0);

  for (int kt = 0; kt < nk; ++kt) {
    if (kt + 1 < nk) {
      issue_tile(kt + 1, (kt + 1) & 1);
      asm volatile("s_wait_asynccnt 0x6" ::: "memory");   // tile kt complete
    } else {
      asm volatile("s_wait_asynccnt 0x0" ::: "memory");
    }
    __syncthreads();                                      // all waves' copies landed

    const u16* Abuf = As + (kt & 1) * AT_ELEM;
    const u16* Bbuf = Bs + (kt & 1) * BT_ELEM;
    #pragma unroll
    for (int kc = 0; kc < 2; ++kc) {                      // two K=32 sub-steps
      FragU bfr[2];
      #pragma unroll
      for (int nt = 0; nt < 2; ++nt) {
        const u16* bp = &Bbuf[(wn * 32 + nt * 16 + row16) * 72 + kc * 32];
        bfr[nt].u[0] = *(const uint4*)(bp + khalf);
        bfr[nt].u[1] = *(const uint4*)(bp + khalf + 16);
      }
      #pragma unroll
      for (int mt = 0; mt < 2; ++mt) {
        FragU a;
        const u16* ap = &Abuf[(wm * 32 + mt * 16 + row16) * 72 + kc * 32];
        a.u[0] = *(const uint4*)(ap + khalf);
        a.u[1] = *(const uint4*)(ap + khalf + 16);
        #pragma unroll
        for (int nt = 0; nt < 2; ++nt) {
          acc[mt][nt] = __builtin_amdgcn_wmma_f32_16x16x32_bf16(
              false, a.bf, false, bfr[nt].bf, (short)0, acc[mt][nt], false, false);
        }
      }
    }
    __syncthreads();   // reads done before this buffer is overwritten by prefetch
  }

  const int hb  = lane >> 4;
  const int col = lane & 15;
  #pragma unroll
  for (int mt = 0; mt < 2; ++mt)
    #pragma unroll
    for (int nt = 0; nt < 2; ++nt)
      #pragma unroll
      for (int r = 0; r < 8; ++r) {
        int mrow = bm + wm * 32 + mt * 16 + r + hb * 8;
        int ncol = bn + wn * 32 + nt * 16 + col;
        size_t idx = (size_t)mrow * N + ncol;
        float v = acc[mt][nt][r];
        if (R) v += R[idx];
        C[idx] = v;
      }
}

// ---------------------------------------------------------------------------
// RoPE on q,k (f32 -> bf16) + V convert with transpose to (head, dim, seq)
// ---------------------------------------------------------------------------
__global__ __launch_bounds__(64)
void rope_cvt(const float* __restrict__ q, const float* __restrict__ k,
              const float* __restrict__ v,
              const float* __restrict__ cos_t, const float* __restrict__ sin_t,
              u16* __restrict__ qb, u16* __restrict__ kb, u16* __restrict__ vT) {
  const int s = blockIdx.x;
  const int h = blockIdx.y;
  const int i = threadIdx.x;            // 0..63
  const size_t base = (size_t)s * DIM + h * HD;
  const float c  = cos_t[s * (HD / 2) + i];
  const float sn = sin_t[s * (HD / 2) + i];

  float q0 = q[base + 2 * i], q1 = q[base + 2 * i + 1];
  qb[base + 2 * i]     = f32_to_bf16(q0 * c - q1 * sn);
  qb[base + 2 * i + 1] = f32_to_bf16(q0 * sn + q1 * c);

  float k0 = k[base + 2 * i], k1 = k[base + 2 * i + 1];
  kb[base + 2 * i]     = f32_to_bf16(k0 * c - k1 * sn);
  kb[base + 2 * i + 1] = f32_to_bf16(k0 * sn + k1 * c);

  vT[((size_t)(h * HD + 2 * i))     * SEQ + s] = f32_to_bf16(v[base + 2 * i]);
  vT[((size_t)(h * HD + 2 * i + 1)) * SEQ + s] = f32_to_bf16(v[base + 2 * i + 1]);
}

// ---------------------------------------------------------------------------
// Flash attention: grid (NH, SEQ/128), 8 waves, 16 query rows per wave.
// QK^T and P*V via WMMA bf16; online softmax; causal mask applied in-kernel.
// ---------------------------------------------------------------------------
__global__ __launch_bounds__(256)
void attn_kernel(const u16* __restrict__ qb, const u16* __restrict__ kb,
                 const u16* __restrict__ vT, u16* __restrict__ ob) {
  __shared__ alignas(16) u16 Plds[8][16 * 40];    // per-wave 16x32 P tile (+pad)
  const int head  = blockIdx.x;
  const int qt    = blockIdx.y;
  const int lane  = threadIdx.x & 31;
  const int wave  = threadIdx.x >> 5;
  const int q0    = qt * 128 + wave * 16;
  const int row16 = lane & 15;
  const int khalf = (lane >> 4) * 8;
  const int hb    = lane >> 4;
  const float scale = 0.08838834764831845f;       // 1/sqrt(128)

  FragU qf[4];
  {
    const u16* qp = qb + (size_t)(q0 + row16) * DIM + head * HD;
    #pragma unroll
    for (int kc = 0; kc < 4; ++kc) {
      qf[kc].u[0] = *(const uint4*)(qp + kc * 32 + khalf);
      qf[kc].u[1] = *(const uint4*)(qp + kc * 32 + khalf + 16);
    }
  }

  v8f o[8] = {};
  float rowM[8], rowL[8];
  #pragma unroll
  for (int r = 0; r < 8; ++r) { rowM[r] = -1e30f; rowL[r] = 0.f; }

  const int jend = qt * 128 + 128;
  for (int j0 = 0; j0 < jend; j0 += 32) {
    v8f sc[2] = {};
    #pragma unroll
    for (int nt = 0; nt < 2; ++nt) {
      const u16* kp = kb + (size_t)(j0 + nt * 16 + row16) * DIM + head * HD;
      #pragma unroll
      for (int kc = 0; kc < 4; ++kc) {
        FragU b;
        b.u[0] = *(const uint4*)(kp + kc * 32 + khalf);
        b.u[1] = *(const uint4*)(kp + kc * 32 + khalf + 16);
        sc[nt] = __builtin_amdgcn_wmma_f32_16x16x32_bf16(
            false, qf[kc].bf, false, b.bf, (short)0, sc[nt], false, false);
      }
    }
    float p0a[8], p1a[8];
    #pragma unroll
    for (int r = 0; r < 8; ++r) {
      const int qi  = q0 + r + hb * 8;
      const int k0i = j0 + (lane & 15);
      const int k1i = j0 + 16 + (lane & 15);
      float s0 = sc[0][r] * scale;
      float s1 = sc[1][r] * scale;
      if (k0i > qi) s0 = -1e30f;
      if (k1i > qi) s1 = -1e30f;
      float mx = fmaxf(s0, s1);
      #pragma unroll
      for (int m = 8; m >= 1; m >>= 1) mx = fmaxf(mx, __shfl_xor(mx, m, 32));
      const float mnew = fmaxf(rowM[r], mx);
      const float corr = __expf(rowM[r] - mnew);
      rowM[r] = mnew;
      float p0 = __expf(s0 - mnew);
      float p1 = __expf(s1 - mnew);
      float rs = p0 + p1;
      #pragma unroll
      for (int m = 8; m >= 1; m >>= 1) rs += __shfl_xor(rs, m, 32);
      rowL[r] = rowL[r] * corr + rs;
      #pragma unroll
      for (int t = 0; t < 8; ++t) o[t][r] *= corr;
      p0a[r] = p0; p1a[r] = p1;
    }
    u16* pl = &Plds[wave][0];
    #pragma unroll
    for (int r = 0; r < 8; ++r) {
      const int m = r + hb * 8;
      pl[m * 40 + (lane & 15)]      = f32_to_bf16(p0a[r]);
      pl[m * 40 + 16 + (lane & 15)] = f32_to_bf16(p1a[r]);
    }
    asm volatile("s_wait_dscnt 0x0" ::: "memory");   // LDS in-order within wave
    FragU pf;
    {
      const u16* pp = &pl[row16 * 40];
      pf.u[0] = *(const uint4*)(pp + khalf);
      pf.u[1] = *(const uint4*)(pp + khalf + 16);
    }
    #pragma unroll
    for (int t = 0; t < 8; ++t) {
      const u16* vp = vT + ((size_t)(head * HD + t * 16 + row16)) * SEQ + j0;
      FragU b;
      b.u[0] = *(const uint4*)(vp + khalf);
      b.u[1] = *(const uint4*)(vp + khalf + 16);
      o[t] = __builtin_amdgcn_wmma_f32_16x16x32_bf16(
          false, pf.bf, false, b.bf, (short)0, o[t], false, false);
    }
  }
  #pragma unroll
  for (int t = 0; t < 8; ++t)
    #pragma unroll
    for (int r = 0; r < 8; ++r) {
      float val = o[t][r] / rowL[r];
      ob[(size_t)(q0 + r + hb * 8) * DIM + head * HD + t * 16 + (lane & 15)] =
          f32_to_bf16(val);
    }
}

// ---------------------------------------------------------------------------
// g = bf16( silu(g1) * g3 )
// ---------------------------------------------------------------------------
__global__ void silu_mul(const float* __restrict__ g1, const float* __restrict__ g3,
                         u16* __restrict__ g, int n) {
  int i = blockIdx.x * blockDim.x + threadIdx.x;
  if (i < n) {
    float a  = g1[i];
    float sv = a / (1.f + __expf(-a));
    g[i] = f32_to_bf16(sv * g3[i]);
  }
}

// ---------------------------------------------------------------------------
extern "C" void kernel_launch(void* const* d_in, const int* in_sizes, int n_in,
                              void* d_out, int out_size, void* d_ws, size_t ws_size,
                              hipStream_t stream) {
  const float* x      = (const float*)d_in[0];
  const float* fcos   = (const float*)d_in[1];
  const float* fsin   = (const float*)d_in[2];
  // d_in[3] = mask: causal, reproduced in-kernel
  const float* anw    = (const float*)d_in[4];
  const float* wq     = (const float*)d_in[5];
  const float* wk     = (const float*)d_in[6];
  const float* wv     = (const float*)d_in[7];
  const float* wo     = (const float*)d_in[8];
  const float* fnw    = (const float*)d_in[9];
  const float* w1     = (const float*)d_in[10];
  const float* w2     = (const float*)d_in[11];
  const float* w3     = (const float*)d_in[12];
  float* out          = (float*)d_out;

  size_t off = 0;
  auto carve = [&](size_t bytes) {
    void* p = (char*)d_ws + off;
    off += (bytes + 255) & ~(size_t)255;
    return p;
  };
  const size_t SD2 = (size_t)SEQ * DIM * 2;
  const size_t SD4 = (size_t)SEQ * DIM * 4;
  const size_t SH4 = (size_t)SEQ * HIDDEN * 4;
  const size_t SH2 = (size_t)SEQ * HIDDEN * 2;
  const size_t WD2 = (size_t)DIM * DIM * 2;
  const size_t WH2 = (size_t)HIDDEN * DIM * 2;

  u16*   xn_bf  = (u16*)  carve(SD2);
  u16*   hn_bf  = (u16*)  carve(SD2);
  float* q_f    = (float*)carve(SD4);            // later reused as g1
  float* k_f    = (float*)carve(SD4);
  float* v_f    = (float*)carve(SD4);
  u16*   q_bf   = (u16*)  carve(SD2);
  u16*   k_bf   = (u16*)  carve(SD2);
  u16*   vT_bf  = (u16*)  carve(SD2);
  u16*   at_bf  = (u16*)  carve(SD2);
  float* h_f    = (float*)carve(SD4);
  float* g3_f   = (float*)carve(SH4);
  u16*   g_bf   = (u16*)  carve(SH2);
  u16*   wq_bf  = (u16*)  carve(WD2);
  u16*   wk_bf  = (u16*)  carve(WD2);
  u16*   wv_bf  = (u16*)  carve(WD2);
  u16*   wo_bf  = (u16*)  carve(WD2);
  u16*   w1_bf  = (u16*)  carve(WH2);
  u16*   w3_bf  = (u16*)  carve(WH2);
  u16*   w2_bf  = (u16*)  carve(WH2);
  float* g1_f   = q_f;   // alias: q/k/v f32 (100.6MB) dead before g1 (90.2MB) lives

  const int nW  = DIM * DIM;
  const int nW1 = HIDDEN * DIM;
  cvt_bf16<<<(nW  / 4 + 255) / 256, 256, 0, stream>>>(wq, wq_bf, nW);
  cvt_bf16<<<(nW  / 4 + 255) / 256, 256, 0, stream>>>(wk, wk_bf, nW);
  cvt_bf16<<<(nW  / 4 + 255) / 256, 256, 0, stream>>>(wv, wv_bf, nW);
  cvt_bf16<<<(nW  / 4 + 255) / 256, 256, 0, stream>>>(wo, wo_bf, nW);
  cvt_bf16<<<(nW1 / 4 + 255) / 256, 256, 0, stream>>>(w1, w1_bf, nW1);
  cvt_bf16<<<(nW1 / 4 + 255) / 256, 256, 0, stream>>>(w3, w3_bf, nW1);
  cvt_bf16<<<(nW1 / 4 + 255) / 256, 256, 0, stream>>>(w2, w2_bf, nW1);

  rmsnorm_bf16<<<SEQ, 256, 0, stream>>>(x, anw, xn_bf);

  dim3 gQ(SEQ / 128, DIM / 64);
  gemm_bf16_nt<<<gQ, 256, 0, stream>>>(xn_bf, wq_bf, q_f, nullptr, SEQ, DIM, DIM);
  gemm_bf16_nt<<<gQ, 256, 0, stream>>>(xn_bf, wk_bf, k_f, nullptr, SEQ, DIM, DIM);
  gemm_bf16_nt<<<gQ, 256, 0, stream>>>(xn_bf, wv_bf, v_f, nullptr, SEQ, DIM, DIM);

  rope_cvt<<<dim3(SEQ, NH), 64, 0, stream>>>(q_f, k_f, v_f, fcos, fsin,
                                             q_bf, k_bf, vT_bf);

  attn_kernel<<<dim3(NH, SEQ / 128), 256, 0, stream>>>(q_bf, k_bf, vT_bf, at_bf);

  gemm_bf16_nt<<<gQ, 256, 0, stream>>>(at_bf, wo_bf, h_f, x, SEQ, DIM, DIM);

  rmsnorm_bf16<<<SEQ, 256, 0, stream>>>(h_f, fnw, hn_bf);

  dim3 gH(SEQ / 128, HIDDEN / 64);
  gemm_bf16_nt<<<gH, 256, 0, stream>>>(hn_bf, w1_bf, g1_f, nullptr, SEQ, HIDDEN, DIM);
  gemm_bf16_nt<<<gH, 256, 0, stream>>>(hn_bf, w3_bf, g3_f, nullptr, SEQ, HIDDEN, DIM);

  const int nG = SEQ * HIDDEN;
  silu_mul<<<(nG + 255) / 256, 256, 0, stream>>>(g1_f, g3_f, g_bf, nG);

  gemm_bf16_nt<<<gQ, 256, 0, stream>>>(g_bf, w2_bf, out, h_f, SEQ, DIM, HIDDEN);
}
